// BitLinear_33552284517049
// MI455X (gfx1250) — 8-bit matrix operand order repaired and checked by reading, compiled, not run on hardware
//
#include <hip/hip_runtime.h>
#include <stdint.h>

typedef __attribute__((ext_vector_type(8))) int v8i;
typedef __attribute__((ext_vector_type(4))) int v4i;

#define M_TOK   16384
#define K_DIM   2048
#define N_DIM   2048
#define W_ELEMS (2048*2048)
#define EPSF    1e-8f

__global__ void bitlin_wabs_sum(const float* __restrict__ w,
                                float* __restrict__ wsum, int n) {
    __shared__ float smem[8];
    float s = 0.0f;
    for (int i = blockIdx.x * blockDim.x + threadIdx.x; i < n;
         i += gridDim.x * blockDim.x) {
        s += fabsf(w[i]);
    }
    #pragma unroll
    for (int off = 16; off > 0; off >>= 1) s += __shfl_xor(s, off, 32);
    const int lane = threadIdx.x & 31, wave = threadIdx.x >> 5;
    if (lane == 0) smem[wave] = s;
    __syncthreads();
    if (threadIdx.x == 0) {
        float t = 0.0f;
        #pragma unroll
        for (int i = 0; i < 8; ++i) t += smem[i];
        atomicAdd(wsum, t);
    }
}

__global__ void bitlin_quant_w(const float* __restrict__ w,
                               const float* __restrict__ wsum,
                               signed char* __restrict__ wq, int n4) {
    const float gamma = wsum[0] * (1.0f / (float)W_ELEMS) + EPSF;
    const float inv   = 1.0f / gamma;
    int i = blockIdx.x * blockDim.x + threadIdx.x;
    if (i >= n4) return;
    const float4 v = ((const float4*)w)[i];
    int q0 = (int)rintf(fminf(1.0f, fmaxf(-1.0f, v.x * inv)));
    int q1 = (int)rintf(fminf(1.0f, fmaxf(-1.0f, v.y * inv)));
    int q2 = (int)rintf(fminf(1.0f, fmaxf(-1.0f, v.z * inv)));
    int q3 = (int)rintf(fminf(1.0f, fmaxf(-1.0f, v.w * inv)));
    uint32_t packed = (uint32_t)(q0 & 0xff) | ((uint32_t)(q1 & 0xff) << 8) |
                      ((uint32_t)(q2 & 0xff) << 16) | ((uint32_t)(q3 & 0xff) << 24);
    ((uint32_t*)wq)[i] = packed;
}

__global__ void bitlin_quant_act(const float* __restrict__ x,
                                 signed char* __restrict__ aq,
                                 float* __restrict__ ascale) {
    __shared__ float smem[8];
    const int row = blockIdx.x;
    const int t   = threadIdx.x;
    const float* xr = x + (size_t)row * K_DIM;

    float v8[8];
    float m = 0.0f;
    #pragma unroll
    for (int j = 0; j < 8; ++j) {
        v8[j] = xr[t + j * 256];
        m = fmaxf(m, fabsf(v8[j]));
    }
    #pragma unroll
    for (int off = 16; off > 0; off >>= 1) m = fmaxf(m, __shfl_xor(m, off, 32));
    const int lane = t & 31, wave = t >> 5;
    if (lane == 0) smem[wave] = m;
    __syncthreads();
    float mm = smem[0];
    #pragma unroll
    for (int i = 1; i < 8; ++i) mm = fmaxf(mm, smem[i]);

    const float scale = mm + EPSF;
    const float inv   = 127.0f / scale;
    signed char* ar = aq + (size_t)row * K_DIM;
    #pragma unroll
    for (int j = 0; j < 8; ++j) {
        float c = fminf(127.0f, fmaxf(-128.0f, v8[j] * inv));
        ar[t + j * 256] = (signed char)(int)rintf(c);
    }
    if (t == 0) ascale[row] = scale;
}

__global__ void __launch_bounds__(256)
bitlin_gemm_iu8(const signed char* __restrict__ aq,
                const signed char* __restrict__ wq,
                const float* __restrict__ ascale,
                const float* __restrict__ wsum,
                float* __restrict__ out) {
    __shared__ signed char ldsB[64 * 64];

    const int t     = threadIdx.x;
    const int lane  = t & 31;
    const int wave  = t >> 5;
    const int r     = lane & 15;
    const int hl    = lane >> 4;

    const int mBase  = blockIdx.y * 256 + wave * 32;
    const int nBase0 = blockIdx.x * 64;

    const int sn   = t >> 2;
    const int skOf = (t & 3) * 16;
    const signed char* wRow = wq + (size_t)(nBase0 + sn) * K_DIM + skOf;

    const signed char* aRow0 = aq + (size_t)(mBase + r)      * K_DIM + hl * 16;
    const signed char* aRow1 = aq + (size_t)(mBase + 16 + r) * K_DIM + hl * 16;

    v8i acc0[4], acc1[4];
    #pragma unroll
    for (int tn = 0; tn < 4; ++tn) { acc0[tn] = (v8i)0; acc1[tn] = (v8i)0; }

    for (int k0 = 0; k0 < K_DIM; k0 += 64) {
        __syncthreads();
        *(v4i*)&ldsB[sn * 64 + skOf] = *(const v4i*)(wRow + k0);
        __syncthreads();

        if (k0 + 64 < K_DIM) {
            __builtin_prefetch(aRow0 + k0 + 64, 0, 1);
            __builtin_prefetch(aRow1 + k0 + 64, 0, 1);
        }

        union { v8i v; uint64_t u[4]; } a0, a1;
        a0.u[0] = *(const uint64_t*)(aRow0 + k0);
        a0.u[1] = *(const uint64_t*)(aRow0 + k0 + 8);
        a0.u[2] = *(const uint64_t*)(aRow0 + k0 + 32);
        a0.u[3] = *(const uint64_t*)(aRow0 + k0 + 40);
        a1.u[0] = *(const uint64_t*)(aRow1 + k0);
        a1.u[1] = *(const uint64_t*)(aRow1 + k0 + 8);
        a1.u[2] = *(const uint64_t*)(aRow1 + k0 + 32);
        a1.u[3] = *(const uint64_t*)(aRow1 + k0 + 40);

        #pragma unroll
        for (int tn = 0; tn < 4; ++tn) {
            const signed char* bp = &ldsB[(tn * 16 + r) * 64 + hl * 16];
            union { v8i v; v4i q[2]; } bf;
            bf.q[0] = *(const v4i*)(bp);
            bf.q[1] = *(const v4i*)(bp + 32);

            acc0[tn] = __builtin_amdgcn_wmma_i32_16x16x64_iu8(
                true, a0.v, true, bf.v, acc0[tn], false, false);
            acc1[tn] = __builtin_amdgcn_wmma_i32_16x16x64_iu8(
                true, a1.v, true, bf.v, acc1[tn], false, false);
        }
    }

    const float gamma = wsum[0] * (1.0f / (float)W_ELEMS) + EPSF;
    float rs0[8], rs1[8];
    #pragma unroll
    for (int v = 0; v < 8; ++v) {
        rs0[v] = gamma * ascale[mBase + v + hl * 8]      * (1.0f / 127.0f);
        rs1[v] = gamma * ascale[mBase + 16 + v + hl * 8] * (1.0f / 127.0f);
    }

    #pragma unroll
    for (int tn = 0; tn < 4; ++tn) {
        const int nCol = nBase0 + tn * 16 + r;
        #pragma unroll
        for (int v = 0; v < 8; ++v) {
            const int row0 = mBase + v + hl * 8;
            out[(size_t)row0 * N_DIM + nCol]        = (float)acc0[tn][v] * rs0[v];
            out[(size_t)(row0 + 16) * N_DIM + nCol] = (float)acc1[tn][v] * rs1[v];
        }
    }
}

extern "C" void kernel_launch(void* const* d_in, const int* in_sizes, int n_in,
                              void* d_out, int out_size, void* d_ws, size_t ws_size,
                              hipStream_t stream) {
    const float* x = (const float*)d_in[0];
    const float* w = (const float*)d_in[1];
    float* out = (float*)d_out;

    char* ws = (char*)d_ws;
    float*       wsum   = (float*)ws;
    float*       ascale = (float*)(ws + 256);
    signed char* wq     = (signed char*)(ws + 256 + 65536 + 256);
    signed char* aq     = (signed char*)(ws + 256 + 65536 + 256 + (size_t)W_ELEMS);

    hipMemsetAsync(wsum, 0, sizeof(float), stream);

    bitlin_wabs_sum<<<512, 256, 0, stream>>>(w, wsum, W_ELEMS);
    bitlin_quant_w<<<(W_ELEMS / 4 + 255) / 256, 256, 0, stream>>>(w, wsum, wq,
                                                                  W_ELEMS / 4);
    bitlin_quant_act<<<M_TOK, 256, 0, stream>>>(x, aq, ascale);

    dim3 grid(N_DIM / 64, M_TOK / 256);
    bitlin_gemm_iu8<<<grid, 256, 0, stream>>>(aq, wq, ascale, wsum, out);
}
